// SparseUNetDecoder_18279380811799
// MI455X (gfx1250) — matrix operand order repaired and checked by reading, not compiled, not run on hardware
//
#include <hip/hip_runtime.h>
#include <hip/hip_bf16.h>

// CDNA5 / gfx1250: wave32, WMMA 16x16x32 bf16 implicit-GEMM U-Net decoder.
// Each wave computes a 16-voxel x 32-outchannel tile (2 WMMA N-tiles) so the
// A (activation) fragment is reused for two v_wmma issues per K-chunk.

typedef __attribute__((ext_vector_type(16))) __bf16 v16bf;
typedef __attribute__((ext_vector_type(8)))  float  v8f;

#define EPSF 1e-5f

__device__ __forceinline__ unsigned short f2bf(float f) {
  unsigned int u = __float_as_uint(f);
  u += 0x7fffu + ((u >> 16) & 1u);          // round-to-nearest-even
  return (unsigned short)(u >> 16);
}

union Frag { v16bf v; uint4 q[2]; };

// ---------------------------------------------------------------------------
// Repack conv weights f32 [t][K][N] -> bf16 WMMA-B fragments
// layout: (((t*KC + kc)*NC + nc)*32 + lane)*16 + h,
// where per ISA 7.12.2 (B 32x16): K = (h>=8?16:0)+(lane>=16?8:0)+h%8, N = nc*16 + lane%16.
// ---------------------------------------------------------------------------
__global__ void pack_b_kernel(const float* __restrict__ src,
                              unsigned short* __restrict__ dst,
                              int NT, int Cin, int Cout) {
  int total = NT * Cin * Cout;
  int idx = blockIdx.x * blockDim.x + threadIdx.x;
  if (idx >= total) return;
  int h    = idx & 15;
  int lane = (idx >> 4) & 31;
  int rest = idx >> 9;
  int NC = Cout >> 4;
  int nc = rest % NC; rest /= NC;
  int KC = Cin >> 5;
  int kc = rest % KC; rest /= KC;
  int t  = rest;
  int K = ((h >= 8) ? 16 : 0) + ((lane >= 16) ? 8 : 0) + (h & 7);
  int N = nc * 16 + (lane & 15);
  float w = src[(size_t)(t * Cin + kc * 32 + K) * Cout + N];
  dst[idx] = f2bf(w);
}

// f32 [V][C] -> bf16 dst[v*stride + choff + c]
__global__ void cvt_bf16_kernel(const float* __restrict__ src,
                                unsigned short* __restrict__ dst,
                                int V, int C, int stride, int choff) {
  size_t total = (size_t)V * C;
  size_t idx = (size_t)blockIdx.x * blockDim.x + threadIdx.x;
  if (idx >= total) return;
  size_t v = idx / C; int c = (int)(idx % C);
  dst[v * (size_t)stride + choff + c] = f2bf(src[idx]);
}

__global__ void zero_kernel(float* __restrict__ p, int n) {
  int i = blockIdx.x * blockDim.x + threadIdx.x;
  if (i < n) p[i] = 0.f;
}

// Masked BN statistics: stats[0..C-1]=sum x*m, [C..2C-1]=sum x^2*m, [2C]=count.
__global__ void bn_stats_kernel(const float* __restrict__ x,
                                const int* __restrict__ mask,
                                float* __restrict__ stats, int V, int C) {
  int c = threadIdx.x;
  float s1 = 0.f, s2 = 0.f, n = 0.f;
  for (int v = blockIdx.x; v < V; v += gridDim.x) {
    if (mask[v] != 0) {
      float f = x[(size_t)v * C + c];
      s1 += f; s2 += f * f; n += 1.f;
    }
  }
  atomicAdd(&stats[c], s1);
  atomicAdd(&stats[C + c], s2);
  if (c == 0) atomicAdd(&stats[2 * C], n);
}

// t = mask ? (x-mean)*g*rsqrt(var+eps)+b : 0 ; t += resid ; t = relu(t)
// writes optional f32 tensor and/or bf16 tensor (strided, for concat packing)
__global__ void bn_apply_kernel(const float* __restrict__ x,
                                const int* __restrict__ mask,
                                const float* __restrict__ g,
                                const float* __restrict__ b,
                                const float* __restrict__ stats,
                                const float* __restrict__ resid,
                                float* __restrict__ outF,
                                unsigned short* __restrict__ outH,
                                int hstride, int choff, int V, int C) {
  size_t total = (size_t)V * C;
  size_t idx = (size_t)blockIdx.x * blockDim.x + threadIdx.x;
  if (idx >= total) return;
  size_t v = idx / C; int c = (int)(idx % C);
  float n = fmaxf(stats[2 * C], 1.f);
  float mean = stats[c] / n;
  float var  = stats[C + c] / n - mean * mean;
  float scale = g[c] * rsqrtf(var + EPSF);
  float t = (mask[v] != 0) ? ((x[idx] - mean) * scale + b[c]) : 0.f;
  if (resid) t += resid[idx];
  t = fmaxf(t, 0.f);
  if (outF) outF[idx] = t;
  if (outH) outH[v * (size_t)hstride + choff + c] = f2bf(t);
}

// ---------------------------------------------------------------------------
// Implicit-GEMM conv (NTAPS = 27 for 3x3x3 SAME, 1 for 1x1x1 projection).
// One wave32 computes a 16-voxel x 32-outchannel tile (two N-tiles share one
// A fragment). K reduced over taps x Cin/32 with v_wmma_f32_16x16x32_bf16.
// A fragment per ISA 7.12.2 (16-bit A 16x32): lane<16 -> K{0..7,16..23},
// lane>=16 -> K{8..15,24..31}; row M = lane%16. OOB taps contribute zeros.
// ---------------------------------------------------------------------------
template <int NTAPS>
__global__ void __launch_bounds__(128)
conv_wmma_kernel(const unsigned short* __restrict__ act,
                 const unsigned short* __restrict__ wpk,
                 float* __restrict__ out, int d, int Cin, int Cout) {
  const int lane = threadIdx.x & 31;
  const int wave = blockIdx.x * (blockDim.x >> 5) + (threadIdx.x >> 5);
  const int NC = Cout >> 4;
  const int NP = NC >> 1;              // N-tile pairs (all Cout here >= 32)
  const int KC = Cin >> 5;
  const int V = d * d * d;
  if (wave >= (V >> 4) * NP) return;
  const int tile = wave / NP;
  const int nc0  = (wave - tile * NP) * 2;
  const int vbase = tile << 4;
  const int v = vbase + (lane & 15);
  const int x = v % d;
  const int y = (v / d) % d;
  const int z = v / (d * d);
  const int koff = (lane >= 16) ? 8 : 0;
  const size_t bstep = (size_t)NC * 32 * 16;

  v8f acc0 = {}, acc1 = {};
  for (int t = 0; t < NTAPS; ++t) {
    int dz = 0, dy = 0, dx = 0;
    if (NTAPS == 27) { dz = t / 9 - 1; dy = (t / 3) % 3 - 1; dx = t % 3 - 1; }
    const int nx = x + dx, ny = y + dy, nz = z + dz;
    const bool valid = (unsigned)nx < (unsigned)d &&
                       (unsigned)ny < (unsigned)d &&
                       (unsigned)nz < (unsigned)d;
    const unsigned short* ap =
        act + (size_t)((nz * d + ny) * d + nx) * Cin + koff;
    const unsigned short* bp =
        wpk + ((((size_t)t * KC) * NC + nc0) * 32 + lane) * 16;
    for (int kc = 0; kc < KC; ++kc) {
      Frag a, b0, b1;
      if (valid) {
        a.q[0] = *(const uint4*)(ap);
        a.q[1] = *(const uint4*)(ap + 16);
      } else {
        a.q[0] = make_uint4(0u, 0u, 0u, 0u);
        a.q[1] = make_uint4(0u, 0u, 0u, 0u);
      }
      b0.q[0] = *(const uint4*)(bp);
      b0.q[1] = *(const uint4*)(bp + 8);
      b1.q[0] = *(const uint4*)(bp + 512);
      b1.q[1] = *(const uint4*)(bp + 520);
      __builtin_prefetch(bp + bstep, 0, 0);
      acc0 = __builtin_amdgcn_wmma_f32_16x16x32_bf16(
          false, a.v, false, b0.v, (short)0, acc0, false, false);
      acc1 = __builtin_amdgcn_wmma_f32_16x16x32_bf16(
          false, a.v, false, b1.v, (short)0, acc1, false, false);
      ap += 32;
      bp += bstep;
    }
  }
  // C/D layout: VGPR r -> M = r + 8*(lane>=16), N = lane%16
  const int col = lane & 15;
  const int hi  = (lane >> 4) & 1;
  float* obase = out + (size_t)(vbase + hi * 8) * Cout + nc0 * 16 + col;
#pragma unroll
  for (int r = 0; r < 8; ++r) {
    obase[(size_t)r * Cout]      = acc0[r];
    obase[(size_t)r * Cout + 16] = acc1[r];
  }
}

// ---------------------------------------------------------------------------
// Transpose-conv 2x2x2 stride 2 (VALID): out[2i+p] = W[p] @ x[i]; 8 parity
// GEMMs, each a 1x1 WMMA GEMM over the coarse grid; dual N-tiles per wave.
// ---------------------------------------------------------------------------
__global__ void __launch_bounds__(128)
upsample_wmma_kernel(const unsigned short* __restrict__ act,
                     const unsigned short* __restrict__ wpk,
                     float* __restrict__ out, int dc, int Cin, int Cout) {
  const int lane = threadIdx.x & 31;
  const int wave = blockIdx.x * (blockDim.x >> 5) + (threadIdx.x >> 5);
  const int NC = Cout >> 4;
  const int NP = NC >> 1;
  const int KC = Cin >> 5;
  const int Vc = dc * dc * dc;
  if (wave >= (Vc >> 4) * NP * 8) return;
  int tmp = wave;
  const int nc0 = (tmp % NP) * 2; tmp /= NP;
  const int par = tmp % 8;        tmp /= 8;
  const int vbase = tmp << 4;
  const int koff = (lane >= 16) ? 8 : 0;
  const size_t bstep = (size_t)NC * 32 * 16;

  const unsigned short* ap = act + (size_t)(vbase + (lane & 15)) * Cin + koff;
  const unsigned short* bp =
      wpk + ((((size_t)par * KC) * NC + nc0) * 32 + lane) * 16;
  v8f acc0 = {}, acc1 = {};
  for (int kc = 0; kc < KC; ++kc) {
    Frag a, b0, b1;
    a.q[0] = *(const uint4*)(ap);
    a.q[1] = *(const uint4*)(ap + 16);
    b0.q[0] = *(const uint4*)(bp);
    b0.q[1] = *(const uint4*)(bp + 8);
    b1.q[0] = *(const uint4*)(bp + 512);
    b1.q[1] = *(const uint4*)(bp + 520);
    acc0 = __builtin_amdgcn_wmma_f32_16x16x32_bf16(
        false, a.v, false, b0.v, (short)0, acc0, false, false);
    acc1 = __builtin_amdgcn_wmma_f32_16x16x32_bf16(
        false, a.v, false, b1.v, (short)0, acc1, false, false);
    ap += 32;
    bp += bstep;
  }
  const int col = lane & 15;
  const int hi  = (lane >> 4) & 1;
  const int px = par & 1, py = (par >> 1) & 1, pz = (par >> 2) & 1;
  const int df = dc * 2;
#pragma unroll
  for (int r = 0; r < 8; ++r) {
    int cv = vbase + hi * 8 + r;
    int cx = cv % dc, cy = (cv / dc) % dc, cz = cv / (dc * dc);
    size_t vf = ((size_t)(2 * cz + pz) * df + (2 * cy + py)) * df + (2 * cx + px);
    out[vf * Cout + nc0 * 16 + col]      = acc0[r];
    out[vf * Cout + nc0 * 16 + 16 + col] = acc1[r];
  }
}

// ---------------------------------------------------------------------------
extern "C" void kernel_launch(void* const* d_in, const int* in_sizes, int n_in,
                              void* d_out, int out_size, void* d_ws, size_t ws_size,
                              hipStream_t stream) {
  (void)in_sizes; (void)out_size; (void)ws_size;
  if (n_in < 74) return;

  const int CUP[4]   = {256, 128, 64, 32};   // decoder out channels per level
  const int CINUP[4] = {512, 256, 128, 64};  // upsample input channels
  const int DF[4]    = {8, 16, 32, 64};      // fine grid dim per level

  const float* enc[5]; const int* msk[5];
  for (int l = 0; l < 5; ++l) {
    enc[l] = (const float*)d_in[2 * l];
    msk[l] = (const int*)d_in[2 * l + 1];
  }
  // pytree leaves: blocks (sorted keys W1,W2,[Wp],b1,b2,g1,g2), then up (W,b,g)
  auto B0 = [&](int i, int k) { return (const float*)d_in[10 + 13 * i + k]; };
  auto B1 = [&](int i, int k) { return (const float*)d_in[10 + 13 * i + 7 + k]; };
  auto UPP = [&](int i, int k) { return (const float*)d_in[62 + 3 * i + k]; };

  char* wsBase = (char*)d_ws; size_t wsOff = 0;
  auto wsAlloc = [&](size_t bytes) -> void* {
    void* p = wsBase + wsOff;
    wsOff = (wsOff + bytes + 255) & ~(size_t)255;
    return p;
  };
  const size_t MAXVC = (size_t)64 * 64 * 64 * 32;  // largest V*C
  float*          BUF  = (float*)wsAlloc(MAXVC * 4);
  float*          RES  = (float*)wsAlloc(MAXVC * 4);
  unsigned short* CAT  = (unsigned short*)wsAlloc(MAXVC * 2 * 2);
  unsigned short* ACT  = (unsigned short*)wsAlloc(MAXVC * 2);
  unsigned short* ACT2 = (unsigned short*)wsAlloc(MAXVC * 2);
  float*          STATS = (float*)wsAlloc(4096);

  auto pack = [&](const float* src, int NT, int Cin, int Cout) -> unsigned short* {
    size_t nEl = (size_t)NT * Cin * Cout;
    unsigned short* dst = (unsigned short*)wsAlloc(nEl * 2);
    pack_b_kernel<<<dim3((unsigned)((nEl + 255) / 256)), dim3(256), 0, stream>>>(
        src, dst, NT, Cin, Cout);
    return dst;
  };
  unsigned short *wup[4], *w1a[4], *w2a[4], *wpp[4], *w1b[4], *w2b[4];
  for (int i = 0; i < 4; ++i) {
    int C = CUP[i], Ci = CINUP[i];
    wup[i] = pack(UPP(i, 0), 8, Ci, C);
    w1a[i] = pack(B0(i, 0), 27, 2 * C, C);
    w2a[i] = pack(B0(i, 1), 27, C, C);
    wpp[i] = pack(B0(i, 2), 1, 2 * C, C);
    w1b[i] = pack(B1(i, 0), 27, C, C);
    w2b[i] = pack(B1(i, 1), 27, C, C);
  }

  // seed coarse activation with enc4 in bf16
  {
    size_t total = (size_t)64 * 512;
    cvt_bf16_kernel<<<dim3((unsigned)((total + 255) / 256)), dim3(256), 0, stream>>>(
        enc[4], ACT, 64, 512, 512, 0);
  }

  auto bn = [&](const float* x, const int* m, const float* g, const float* bb,
                const float* resid, float* outF, unsigned short* outH,
                int hs, int ho, int V, int C) {
    int n = 2 * C + 1;
    zero_kernel<<<dim3((unsigned)((n + 255) / 256)), dim3(256), 0, stream>>>(STATS, n);
    int grid = V < 2048 ? V : 2048;
    bn_stats_kernel<<<dim3((unsigned)grid), dim3((unsigned)C), 0, stream>>>(
        x, m, STATS, V, C);
    size_t total = (size_t)V * C;
    bn_apply_kernel<<<dim3((unsigned)((total + 255) / 256)), dim3(256), 0, stream>>>(
        x, m, g, bb, STATS, resid, outF, outH, hs, ho, V, C);
  };
  auto conv27 = [&](const unsigned short* a, const unsigned short* w, float* o,
                    int d, int Ci, int Co) {
    int waves = (d * d * d / 16) * (Co / 32);
    conv_wmma_kernel<27><<<dim3((unsigned)((waves + 3) / 4)), dim3(128), 0, stream>>>(
        a, w, o, d, Ci, Co);
  };
  auto conv1t = [&](const unsigned short* a, const unsigned short* w, float* o,
                    int d, int Ci, int Co) {
    int waves = (d * d * d / 16) * (Co / 32);
    conv_wmma_kernel<1><<<dim3((unsigned)((waves + 3) / 4)), dim3(128), 0, stream>>>(
        a, w, o, d, Ci, Co);
  };

  for (int i = 0; i < 4; ++i) {
    int C = CUP[i], Ci = CINUP[i], d = DF[i], dc = d / 2;
    int V = d * d * d, Vc = dc * dc * dc;
    const int* mf = msk[3 - i];

    // upsample (transpose conv) ACT(coarse,Ci) -> BUF f32 (fine,C)
    {
      int waves = (Vc / 16) * (C / 32) * 8;
      upsample_wmma_kernel<<<dim3((unsigned)((waves + 3) / 4)), dim3(128), 0, stream>>>(
          ACT, wup[i], BUF, dc, Ci, C);
    }
    // BN+ReLU -> first half of concat (bf16)
    bn(BUF, mf, UPP(i, 2), UPP(i, 1), nullptr, nullptr, CAT, 2 * C, 0, V, C);
    // skip -> second half of concat (bf16)
    {
      size_t total = (size_t)V * C;
      cvt_bf16_kernel<<<dim3((unsigned)((total + 255) / 256)), dim3(256), 0, stream>>>(
          enc[3 - i], CAT, V, C, 2 * C, C);
    }
    // ---- resblock A (2C -> C, with 1x1x1 projection)
    conv27(CAT, w1a[i], BUF, d, 2 * C, C);
    conv1t(CAT, wpp[i], RES, d, 2 * C, C);             // identity branch
    bn(BUF, mf, B0(i, 5), B0(i, 3), nullptr, nullptr, ACT, C, 0, V, C);
    conv27(ACT, w2a[i], BUF, d, C, C);
    bn(BUF, mf, B0(i, 6), B0(i, 4), RES, RES, ACT, C, 0, V, C);
    // ---- resblock B (C -> C, identity = RES)
    conv27(ACT, w1b[i], BUF, d, C, C);
    bn(BUF, mf, B1(i, 4), B1(i, 2), nullptr, nullptr, ACT2, C, 0, V, C);
    conv27(ACT2, w2b[i], BUF, d, C, C);
    if (i < 3)
      bn(BUF, mf, B1(i, 5), B1(i, 3), RES, nullptr, ACT, C, 0, V, C);
    else
      bn(BUF, mf, B1(i, 5), B1(i, 3), RES, (float*)d_out, nullptr, 0, 0, V, C);
  }
}
